// PosDepMorphWordModel_20890720927784
// MI455X (gfx1250) — compile-verified, matrix operand order checked by reading, not run on hardware
//
#include <hip/hip_runtime.h>
#include <hip/hip_bf16.h>
#include <hip/hip_fp16.h>

typedef __attribute__((ext_vector_type(16))) _Float16 v16h;
typedef __attribute__((ext_vector_type(8)))  _Float16 v8h;
typedef __attribute__((ext_vector_type(8)))  float    v8f;
typedef __attribute__((ext_vector_type(4)))  int      v4i;

#define TT    128
#define BB    32
#define HID   200
#define GATES 800
#define HKP   224     // HID padded to mult of 32
#define HKPP  240     // HKP + 16 pad (LDS bank stagger)
#define HCAT  416     // 400 (and 401) padded to mult of 32
#define EMBP  320     // 300 padded
#define NRELC 50
#define TP1   129
#define RELW  (NRELC*HCAT)   // 20800
#define NPM   66      // 18 pos + 4*12 morph
#define BM    64
#define BN    256
#define KMAX  416
#define KPADL (KMAX + 16)    // LDS A-panel row stride

#define AS1 __attribute__((address_space(1)))
#define AS3 __attribute__((address_space(3)))

#if defined(__AMDGCN__) && __has_builtin(__builtin_amdgcn_global_load_async_to_lds_b128)
#define HAVE_ASYNC_LDS 1
#else
#define HAVE_ASYNC_LDS 0
#endif
#if defined(__AMDGCN__) && __has_builtin(__builtin_amdgcn_s_wait_asynccnt)
#define HAVE_WAIT_ASYNC 1
#else
#define HAVE_WAIT_ASYNC 0
#endif

__device__ __forceinline__ void cp_lds_b128(_Float16* ldst, const _Float16* gsrc) {
#if HAVE_ASYNC_LDS
    __builtin_amdgcn_global_load_async_to_lds_b128(
        (AS1 v4i*)gsrc, (AS3 v4i*)ldst, 0, 0);
#else
    *(v8h*)ldst = *(const v8h*)gsrc;
#endif
}

__device__ __forceinline__ void wait_cp_lds() {
#if HAVE_ASYNC_LDS
#if HAVE_WAIT_ASYNC
    __builtin_amdgcn_s_wait_asynccnt(0);
#else
    asm volatile("s_wait_asynccnt 0" ::: "memory");
#endif
#endif
}

// ---------------------------------------------------------------------------
// WMMA GEMM:  C[M,N] = A[M,K] * Bt[N,K]^T  (+bias, relu)
// K multiple of 32, K <= KMAX. Block: 256 threads = 8 waves as 2(M) x 4(N);
// block tile 64x256, wave tile 32x64: 2 A fragments (LDS) x 4 B fragments
// (global), 8 WMMAs per K-step -> 1 global b128 per WMMA.
// A panel [64][K] staged in LDS via async-to-LDS. z-batched: A += z*sAc,
// Cf += z*sCc.
// ---------------------------------------------------------------------------
__global__ __launch_bounds__(256) void gemm_wmma(
    const _Float16* __restrict__ A,  long lda, long sAc,
    const _Float16* __restrict__ Bt, long ldb,
    float* __restrict__ Cf, long ldc, long sCc,
    _Float16* __restrict__ Ch, long ldch,
    const float* __restrict__ bias, int relu,
    int M, int N, int K)
{
    __shared__ alignas(64) _Float16 Ap[BM][KPADL];   // 55296 B

    int z = blockIdx.z;
    A += (long)z * sAc;
    if (Cf) Cf += (long)z * sCc;

    int mbase = blockIdx.x * BM;

    // cooperative A-panel load: 16-byte chunks, rows clamped at M-1
    int kch = K >> 3;
    for (int c = threadIdx.x; c < BM * kch; c += 256) {
        int r = c / kch, kc = (c - r * kch) << 3;
        int gr = mbase + r; if (gr >= M) gr = M - 1;
        cp_lds_b128(&Ap[r][kc], A + (long)gr * lda + kc);
    }
    wait_cp_lds();
    __syncthreads();

    int lane = threadIdx.x & 31;
    int wave = threadIdx.x >> 5;
    int wm = wave & 1, wn = wave >> 1;   // 2 x 4 wave grid
    int tm = mbase + wm * 32;
    int tn = blockIdx.y * BN + wn * 64;
    if (tm >= M || tn >= N) return;      // uniform per wave; no barriers below

    int koff = (lane >> 4) * 16;
    const _Float16* arow0 = &Ap[wm * 32 + (lane & 15)][koff];
    const _Float16* arow1 = &Ap[wm * 32 + 16 + (lane & 15)][koff];
    const _Float16* brow[4];
#pragma unroll
    for (int i = 0; i < 4; ++i) {
        int bn = tn + i * 16 + (lane & 15); if (bn >= N) bn = N - 1;
        brow[i] = Bt + (long)bn * ldb + koff;
    }

    v8f acc[2][4] = {};
    for (int k0 = 0; k0 < K; k0 += 32) {
        v16h av0 = *(const v16h*)(arow0 + k0);
        v16h av1 = *(const v16h*)(arow1 + k0);
#pragma unroll
        for (int i = 0; i < 4; ++i) {
            v16h bv = *(const v16h*)(brow[i] + k0);
            acc[0][i] = __builtin_amdgcn_wmma_f32_16x16x32_f16(
                false, av0, false, bv, (short)0, acc[0][i], false, false);
            acc[1][i] = __builtin_amdgcn_wmma_f32_16x16x32_f16(
                false, av1, false, bv, (short)0, acc[1][i], false, false);
        }
    }

    int mhi = 8 * (lane >> 4);
#pragma unroll
    for (int i = 0; i < 4; ++i) {
        int nn = tn + i * 16 + (lane & 15);
        if (nn >= N) continue;
        float badd = bias ? bias[nn] : 0.f;
#pragma unroll
        for (int mi = 0; mi < 2; ++mi) {
#pragma unroll
            for (int r = 0; r < 8; ++r) {
                int m = tm + mi * 16 + r + mhi;
                if (m >= M) continue;
                float v = acc[mi][i][r] + badd;
                if (relu && v < 0.f) v = 0.f;
                if (Cf) Cf[(long)m * ldc + nn] = v;
                if (Ch) Ch[(long)m * ldch + nn] = (_Float16)v;
            }
        }
    }
}

static inline dim3 gemm_grid(int M, int N, int z) {
    return dim3((M + BM - 1) / BM, (N + BN - 1) / BN, z);
}

// ---------------------------------------------------------------------------
// Persistent BiLSTM direction kernel. grid.x = 2 (0=forward, 1=backward).
// Per step: g = h @ Whh^T via WMMA (LDS h, global f16 Whh), then gate math.
// 800 threads = 25 waves; each wave owns a 32(M) x 32(N) tile: 2 A fragments
// shared across 2 B fragments -> 4 WMMAs, 1 global b128 per WMMA.
// ---------------------------------------------------------------------------
__device__ __forceinline__ float sigf(float x) { return 1.f / (1.f + __expf(-x)); }

__global__ __launch_bounds__(800) void lstm_layer(
    const float* __restrict__ xWf, const float* __restrict__ xWb,
    const _Float16* __restrict__ Whhf, const _Float16* __restrict__ Whhb,
    const float* __restrict__ bihf, const float* __restrict__ bhhf,
    const float* __restrict__ bihb, const float* __restrict__ bhhb,
    _Float16* __restrict__ hs)     // [T*B][HCAT], fwd cols 0..199, bwd 200..399
{
    __shared__ alignas(64) _Float16 hbuf[BB][HKPP];    // 15360 B
    __shared__ alignas(64) _Float16 gbuf[BB][GATES];   // 51200 B

    int dir = blockIdx.x;
    const float*    xW   = dir ? xWb  : xWf;
    const _Float16* Whh  = dir ? Whhb : Whhf;
    const float*    bih  = dir ? bihb : bihf;
    const float*    bhh  = dir ? bhhb : bhhf;
    int colbase = dir ? HID : 0;

    int tid  = threadIdx.x;
    int lane = tid & 31;
    int wave = tid >> 5;    // 25 waves

    for (int i = tid; i < BB * HKPP; i += 800) ((_Float16*)hbuf)[i] = (_Float16)0.f;
    float cstate[8];
#pragma unroll
    for (int q = 0; q < 8; ++q) cstate[q] = 0.f;
    __syncthreads();

    int koff = (lane >> 4) * 16;
    int mhi = 8 * (lane >> 4);
    int nbase = wave * 32;                 // 25 waves x 32 cols = 800
    const _Float16* arow0 = &hbuf[(lane & 15)][koff];
    const _Float16* arow1 = &hbuf[16 + (lane & 15)][koff];
    const _Float16* brow0 = Whh + (long)(nbase + (lane & 15)) * HKP + koff;
    const _Float16* brow1 = Whh + (long)(nbase + 16 + (lane & 15)) * HKP + koff;

    for (int step = 0; step < TT; ++step) {
        int t = dir ? (TT - 1 - step) : step;
        v8f acc[2][2] = {};
#pragma unroll
        for (int k0 = 0; k0 < HKP; k0 += 32) {
            v16h av0 = *(const v16h*)(arow0 + k0);
            v16h av1 = *(const v16h*)(arow1 + k0);
            v16h bv0 = *(const v16h*)(brow0 + k0);
            v16h bv1 = *(const v16h*)(brow1 + k0);
            acc[0][0] = __builtin_amdgcn_wmma_f32_16x16x32_f16(
                false, av0, false, bv0, (short)0, acc[0][0], false, false);
            acc[1][0] = __builtin_amdgcn_wmma_f32_16x16x32_f16(
                false, av1, false, bv0, (short)0, acc[1][0], false, false);
            acc[0][1] = __builtin_amdgcn_wmma_f32_16x16x32_f16(
                false, av0, false, bv1, (short)0, acc[0][1], false, false);
            acc[1][1] = __builtin_amdgcn_wmma_f32_16x16x32_f16(
                false, av1, false, bv1, (short)0, acc[1][1], false, false);
        }
#pragma unroll
        for (int mi = 0; mi < 2; ++mi)
#pragma unroll
            for (int j = 0; j < 2; ++j)
#pragma unroll
                for (int r = 0; r < 8; ++r)
                    gbuf[mi * 16 + r + mhi][nbase + j * 16 + (lane & 15)] =
                        (_Float16)acc[mi][j][r];
        __syncthreads();

        const float* xw = xW + (long)t * BB * GATES;
#pragma unroll
        for (int q = 0; q < 8; ++q) {
            int s = tid + q * 800;           // 0..6399
            int b = s / HID, j = s % HID;
            long xr = (long)b * GATES;
            float gi = (float)gbuf[b][j]       + xw[xr + j]       + bih[j]       + bhh[j];
            float gf = (float)gbuf[b][HID + j] + xw[xr + HID + j] + bih[HID + j] + bhh[HID + j];
            float gg = (float)gbuf[b][400 + j] + xw[xr + 400 + j] + bih[400 + j] + bhh[400 + j];
            float go = (float)gbuf[b][600 + j] + xw[xr + 600 + j] + bih[600 + j] + bhh[600 + j];
            float c = sigf(gf) * cstate[q] + sigf(gi) * tanhf(gg);
            cstate[q] = c;
            float h = sigf(go) * tanhf(c);
            hbuf[b][j] = (_Float16)h;
            hs[((long)t * BB + b) * HCAT + colbase + j] = (_Float16)h;
        }
        __syncthreads();
    }
}

// ---------------------------------------------------------------------------
// Prep / epilogue kernels
// ---------------------------------------------------------------------------
__global__ void embed_gather(const int* __restrict__ x, const float* __restrict__ E,
                             _Float16* __restrict__ e)
{
    long i = (long)blockIdx.x * blockDim.x + threadIdx.x;   // T*B*EMBP
    if (i >= (long)TT * BB * EMBP) return;
    int col = i % EMBP; long row = i / EMBP;
    int b = row % BB, t = row / BB;
    float v = 0.f;
    if (col < 300) { int tok = x[b * TT + t]; v = E[(long)tok * 300 + col]; }
    e[row * EMBP + col] = (_Float16)v;
}

__global__ void cvt_pad(const float* __restrict__ in, _Float16* __restrict__ out,
                        int R, int C, int CP)
{
    long i = (long)blockIdx.x * blockDim.x + threadIdx.x;
    if (i >= (long)R * CP) return;
    int c = i % CP; long r = i / CP;
    out[i] = (_Float16)((c < C) ? in[r * C + c] : 0.f);
}

// out[n][k] = in[k][n] (in: [K0][N0]), k padded to KP
__global__ void cvt_padT(const float* __restrict__ in, _Float16* __restrict__ out,
                         int K0, int N0, int KP)
{
    long i = (long)blockIdx.x * blockDim.x + threadIdx.x;
    if (i >= (long)N0 * KP) return;
    int k = i % KP; long n = i / KP;
    out[i] = (_Float16)((k < K0) ? in[(long)k * N0 + n] : 0.f);
}

__global__ void uarcT_build(const float* __restrict__ U, _Float16* __restrict__ out)
{
    long i = (long)blockIdx.x * blockDim.x + threadIdx.x;   // 416*416
    if (i >= (long)HCAT * HCAT) return;
    int ii = i % HCAT; long j = i / HCAT;
    out[i] = (_Float16)((ii < 401 && j < 401) ? U[(long)ii * 401 + j] : 0.f);
}

__global__ void urelT_build(const float* __restrict__ U, _Float16* __restrict__ out)
{
    long i = (long)blockIdx.x * blockDim.x + threadIdx.x;   // 20800*416
    if (i >= (long)RELW * HCAT) return;
    int ii = i % HCAT; long row = i / HCAT;
    int c = row / HCAT, j = row % HCAT;
    float v = 0.f;
    if (ii < 401 && j < 401) v = U[((long)ii * NRELC + c) * 401 + j];
    out[i] = (_Float16)v;
}

__global__ void wpm_build(const float* __restrict__ Wpos, const float* __restrict__ Wm,
                          const float* __restrict__ bpos, const float* __restrict__ bm,
                          _Float16* __restrict__ Wt, float* __restrict__ bias)
{
    long i = (long)blockIdx.x * blockDim.x + threadIdx.x;   // 80*416
    if (i >= 80L * HCAT) return;
    int k = i % HCAT; long n = i / HCAT;
    float v = 0.f;
    if (k < 400) {
        if (n < 18) v = Wpos[(long)k * 18 + n];
        else if (n < NPM) { int a = (n - 18) / 12, vv = (n - 18) % 12; v = Wm[((long)a * 400 + k) * 12 + vv]; }
    }
    Wt[i] = (_Float16)v;
    if (k == 0 && n < NPM) bias[n] = (n < 18) ? bpos[n] : bm[(n - 18)];
}

__global__ void init_pad(_Float16* __restrict__ p, long n, int ld, int onecol)
{
    long i = (long)blockIdx.x * blockDim.x + threadIdx.x;
    if (i >= n) return;
    int c = i % ld;
    p[i] = (_Float16)((c == onecol) ? 1.f : 0.f);
}

__global__ void rooted_build(const _Float16* __restrict__ hs, const float* __restrict__ root,
                             _Float16* __restrict__ out)
{
    long idx = (long)blockIdx.x * blockDim.x + threadIdx.x;  // 4128*416
    if (idx >= (long)BB * TP1 * HCAT) return;
    int col = idx % HCAT; long row = idx / HCAT;
    int b = row / TP1, i = row % TP1;
    float v;
    if (i == 0) v = (col < 400) ? root[col] : 0.f;
    else        v = (float)hs[((long)(i - 1) * BB + b) * HCAT + col];
    out[idx] = (_Float16)v;
}

__global__ void pm_scatter(const float* __restrict__ pm, const int* __restrict__ mask,
                           float* __restrict__ pos, float* __restrict__ morph)
{
    long i = (long)blockIdx.x * blockDim.x + threadIdx.x;   // 4096*66
    if (i >= 4096L * NPM) return;
    int n = i % NPM; long row = i / NPM;
    int b = row % BB, t = row / BB;
    float m = (float)mask[b * TT + t];
    float v = pm[row * NPM + n] * m;
    if (n < 18) pos[((long)b * TT + t) * 18 + n] = v;
    else { int a = (n - 18) / 12, vv = (n - 18) % 12; morph[(((long)b * TT + t) * 4 + a) * 12 + vv] = v; }
}

__global__ void arc_scatter(const float* __restrict__ arcfull, const int* __restrict__ mask,
                            float* __restrict__ arc)
{
    long i = (long)blockIdx.x * blockDim.x + threadIdx.x;   // 32*128*129
    if (i >= (long)BB * TT * TP1) return;
    int k = i % TP1; long r = i / TP1;
    int t = r % TT, b = r / TT;
    float m  = (float)mask[b * TT + t];
    float rm = (k == 0) ? 1.f : (float)mask[b * TT + (k - 1)];
    arc[i] = arcfull[((long)b * TP1 + (t + 1)) * TP1 + k] * m * rm;
}

__global__ void rel_mask(float* __restrict__ rel, const int* __restrict__ mask)
{
    long i = (long)blockIdx.x * blockDim.x + threadIdx.x;   // 32*129*50*129
    if (i >= (long)BB * TP1 * NRELC * TP1) return;
    int k = i % TP1; long r = i / TP1;
    r /= NRELC;
    int t = r % TP1, b = r / TP1;
    float rmt = (t == 0) ? 1.f : (float)mask[b * TT + (t - 1)];
    float rmk = (k == 0) ? 1.f : (float)mask[b * TT + (k - 1)];
    rel[i] *= rmt * rmk;
}

// ---------------------------------------------------------------------------
static inline char* bump(char*& p, size_t bytes) {
    char* r = p; p += (bytes + 255) & ~(size_t)255; return r;
}
#define G1(n) dim3((unsigned)(((n) + 255) / 256)), dim3(256)

extern "C" void kernel_launch(void* const* d_in, const int* in_sizes, int n_in,
                              void* d_out, int out_size, void* d_ws, size_t ws_size,
                              hipStream_t stream) {
    const int*   x    = (const int*)  d_in[0];
    const int*   mask = (const int*)  d_in[1];
    const float* E    = (const float*)d_in[2];
    const float* Wih[4] = {(const float*)d_in[3],  (const float*)d_in[7],
                           (const float*)d_in[11], (const float*)d_in[15]};
    const float* Whh[4] = {(const float*)d_in[4],  (const float*)d_in[8],
                           (const float*)d_in[12], (const float*)d_in[16]};
    const float* bih[4] = {(const float*)d_in[5],  (const float*)d_in[9],
                           (const float*)d_in[13], (const float*)d_in[17]};
    const float* bhh[4] = {(const float*)d_in[6],  (const float*)d_in[10],
                           (const float*)d_in[14], (const float*)d_in[18]};
    const float* W_pos = (const float*)d_in[19]; const float* b_pos = (const float*)d_in[20];
    const float* W_ad  = (const float*)d_in[21]; const float* b_ad  = (const float*)d_in[22];
    const float* W_ah  = (const float*)d_in[23]; const float* b_ah  = (const float*)d_in[24];
    const float* W_rd  = (const float*)d_in[25]; const float* b_rd  = (const float*)d_in[26];
    const float* W_rh  = (const float*)d_in[27]; const float* b_rh  = (const float*)d_in[28];
    const float* U_arc = (const float*)d_in[29]; const float* U_rel = (const float*)d_in[30];
    const float* root  = (const float*)d_in[31];
    const float* W_mo  = (const float*)d_in[32]; const float* b_mo  = (const float*)d_in[33];

    float* out_pos = (float*)d_out;
    float* out_arc = out_pos + (long)BB * TT * 18;
    float* out_rel = out_arc + (long)BB * TT * TP1;
    float* out_mor = out_rel + (long)BB * TP1 * NRELC * TP1;

    char* wp = (char*)d_ws;
    _Float16* e_h    = (_Float16*)bump(wp, 4096L * EMBP * 2);
    _Float16* WihH[4]; _Float16* WhhH[4];
    int dinp[4] = {EMBP, EMBP, HCAT, HCAT};
    for (int l = 0; l < 4; ++l) {
        WihH[l] = (_Float16*)bump(wp, (long)GATES * dinp[l] * 2);
        WhhH[l] = (_Float16*)bump(wp, (long)GATES * HKP * 2);
    }
    float* xWf = (float*)bump(wp, 4096L * GATES * 4);
    float* xWb = (float*)bump(wp, 4096L * GATES * 4);
    _Float16* hs0 = (_Float16*)bump(wp, 4096L * HCAT * 2);
    _Float16* hs1 = (_Float16*)bump(wp, 4096L * HCAT * 2);
    _Float16* WadT = (_Float16*)bump(wp, 400L * HCAT * 2);
    _Float16* WahT = (_Float16*)bump(wp, 400L * HCAT * 2);
    _Float16* WrdT = (_Float16*)bump(wp, 400L * HCAT * 2);
    _Float16* WrhT = (_Float16*)bump(wp, 400L * HCAT * 2);
    _Float16* WpmT = (_Float16*)bump(wp, 80L * HCAT * 2);
    float*    bpm  = (float*)   bump(wp, NPM * 4);
    float*    pm   = (float*)   bump(wp, 4096L * NPM * 4);
    _Float16* rooted = (_Float16*)bump(wp, (long)BB * TP1 * HCAT * 2);
    _Float16* da = (_Float16*)bump(wp, (long)BB * TP1 * HCAT * 2);
    _Float16* ha = (_Float16*)bump(wp, (long)BB * TP1 * HCAT * 2);
    _Float16* dr = (_Float16*)bump(wp, (long)BB * TP1 * HCAT * 2);
    _Float16* hr = (_Float16*)bump(wp, (long)BB * TP1 * HCAT * 2);
    _Float16* UarcT = (_Float16*)bump(wp, (long)HCAT * HCAT * 2);
    _Float16* UrelT = (_Float16*)bump(wp, (long)RELW * HCAT * 2);
    _Float16* arcT  = (_Float16*)bump(wp, (long)BB * TP1 * HCAT * 2);
    float*    arcfull = (float*)bump(wp, (long)BB * TP1 * TP1 * 4);
    _Float16* chunk = (_Float16*)bump(wp, (long)TP1 * RELW * 2);

    // ---- weight prep ----
    int dins[4] = {300, 300, 400, 400};
    for (int l = 0; l < 4; ++l) {
        cvt_pad<<<G1((long)GATES * dinp[l]), 0, stream>>>(Wih[l], WihH[l], GATES, dins[l], dinp[l]);
        cvt_pad<<<G1((long)GATES * HKP),     0, stream>>>(Whh[l], WhhH[l], GATES, HID, HKP);
    }
    cvt_padT<<<G1(400L * HCAT), 0, stream>>>(W_ad, WadT, 400, 400, HCAT);
    cvt_padT<<<G1(400L * HCAT), 0, stream>>>(W_ah, WahT, 400, 400, HCAT);
    cvt_padT<<<G1(400L * HCAT), 0, stream>>>(W_rd, WrdT, 400, 400, HCAT);
    cvt_padT<<<G1(400L * HCAT), 0, stream>>>(W_rh, WrhT, 400, 400, HCAT);
    uarcT_build<<<G1((long)HCAT * HCAT), 0, stream>>>(U_arc, UarcT);
    urelT_build<<<G1((long)RELW * HCAT), 0, stream>>>(U_rel, UrelT);
    wpm_build<<<G1(80L * HCAT), 0, stream>>>(W_pos, W_mo, b_pos, b_mo, WpmT, bpm);
    init_pad<<<G1(4096L * HCAT), 0, stream>>>(hs0, 4096L * HCAT, HCAT, -1);
    init_pad<<<G1(4096L * HCAT), 0, stream>>>(hs1, 4096L * HCAT, HCAT, -1);
    init_pad<<<G1((long)BB * TP1 * HCAT), 0, stream>>>(da, (long)BB * TP1 * HCAT, HCAT, 400);
    init_pad<<<G1((long)BB * TP1 * HCAT), 0, stream>>>(ha, (long)BB * TP1 * HCAT, HCAT, 400);
    init_pad<<<G1((long)BB * TP1 * HCAT), 0, stream>>>(dr, (long)BB * TP1 * HCAT, HCAT, 400);
    init_pad<<<G1((long)BB * TP1 * HCAT), 0, stream>>>(hr, (long)BB * TP1 * HCAT, HCAT, 400);
    init_pad<<<G1((long)BB * TP1 * HCAT), 0, stream>>>(arcT, (long)BB * TP1 * HCAT, HCAT, -1);

    // ---- embedding ----
    embed_gather<<<G1(4096L * EMBP), 0, stream>>>(x, E, e_h);

    // ---- layer 0: input projection + recurrence ----
    gemm_wmma<<<gemm_grid(4096, GATES, 1), 256, 0, stream>>>(e_h, EMBP, 0, WihH[0], EMBP,
        xWf, GATES, 0, nullptr, 0, nullptr, 0, 4096, GATES, EMBP);
    gemm_wmma<<<gemm_grid(4096, GATES, 1), 256, 0, stream>>>(e_h, EMBP, 0, WihH[1], EMBP,
        xWb, GATES, 0, nullptr, 0, nullptr, 0, 4096, GATES, EMBP);
    lstm_layer<<<dim3(2), dim3(800), 0, stream>>>(xWf, xWb, WhhH[0], WhhH[1],
        bih[0], bhh[0], bih[1], bhh[1], hs0);

    // ---- layer 1 ----
    gemm_wmma<<<gemm_grid(4096, GATES, 1), 256, 0, stream>>>(hs0, HCAT, 0, WihH[2], HCAT,
        xWf, GATES, 0, nullptr, 0, nullptr, 0, 4096, GATES, HCAT);
    gemm_wmma<<<gemm_grid(4096, GATES, 1), 256, 0, stream>>>(hs0, HCAT, 0, WihH[3], HCAT,
        xWb, GATES, 0, nullptr, 0, nullptr, 0, 4096, GATES, HCAT);
    lstm_layer<<<dim3(2), dim3(800), 0, stream>>>(xWf, xWb, WhhH[2], WhhH[3],
        bih[2], bhh[2], bih[3], bhh[3], hs1);

    // ---- pos + morph heads ----
    gemm_wmma<<<gemm_grid(4096, NPM, 1), 256, 0, stream>>>(hs1, HCAT, 0, WpmT, HCAT,
        pm, NPM, 0, nullptr, 0, bpm, 0, 4096, NPM, HCAT);
    pm_scatter<<<G1(4096L * NPM), 0, stream>>>(pm, mask, out_pos, out_mor);

    // ---- rooted + MLP heads (relu, f16 out with ones-feature col 400) ----
    rooted_build<<<G1((long)BB * TP1 * HCAT), 0, stream>>>(hs1, root, rooted);
    gemm_wmma<<<gemm_grid(BB * TP1, 400, 1), 256, 0, stream>>>(rooted, HCAT, 0, WadT, HCAT,
        nullptr, 0, 0, da, HCAT, b_ad, 1, BB * TP1, 400, HCAT);
    gemm_wmma<<<gemm_grid(BB * TP1, 400, 1), 256, 0, stream>>>(rooted, HCAT, 0, WahT, HCAT,
        nullptr, 0, 0, ha, HCAT, b_ah, 1, BB * TP1, 400, HCAT);
    gemm_wmma<<<gemm_grid(BB * TP1, 400, 1), 256, 0, stream>>>(rooted, HCAT, 0, WrdT, HCAT,
        nullptr, 0, 0, dr, HCAT, b_rd, 1, BB * TP1, 400, HCAT);
    gemm_wmma<<<gemm_grid(BB * TP1, 400, 1), 256, 0, stream>>>(rooted, HCAT, 0, WrhT, HCAT,
        nullptr, 0, 0, hr, HCAT, b_rh, 1, BB * TP1, 400, HCAT);

    // ---- arc biaffine ----
    gemm_wmma<<<gemm_grid(BB * TP1, 401, 1), 256, 0, stream>>>(da, HCAT, 0, UarcT, HCAT,
        nullptr, 0, 0, arcT, HCAT, nullptr, 0, BB * TP1, 401, HCAT);
    for (int b = 0; b < BB; ++b) {
        gemm_wmma<<<gemm_grid(TP1, TP1, 1), 256, 0, stream>>>(
            arcT + (long)b * TP1 * HCAT, HCAT, 0,
            ha   + (long)b * TP1 * HCAT, HCAT,
            arcfull + (long)b * TP1 * TP1, TP1, 0,
            nullptr, 0, nullptr, 0, TP1, TP1, HCAT);
    }
    arc_scatter<<<G1((long)BB * TT * TP1), 0, stream>>>(arcfull, mask, out_arc);

    // ---- rel biaffine (chunked per batch) ----
    for (int b = 0; b < BB; ++b) {
        gemm_wmma<<<gemm_grid(TP1, RELW, 1), 256, 0, stream>>>(
            dr + (long)b * TP1 * HCAT, HCAT, 0, UrelT, HCAT,
            nullptr, 0, 0, chunk, RELW, nullptr, 0, TP1, RELW, HCAT);
        gemm_wmma<<<gemm_grid(TP1, TP1, NRELC), 256, 0, stream>>>(
            chunk, RELW, HCAT,
            hr + (long)b * TP1 * HCAT, HCAT,
            out_rel + (long)b * TP1 * NRELC * TP1, (long)NRELC * TP1, TP1,
            nullptr, 0, nullptr, 0, TP1, TP1, HCAT);
    }
    rel_mask<<<G1((long)BB * TP1 * NRELC * TP1), 0, stream>>>(out_rel, mask);

    (void)in_sizes; (void)n_in; (void)out_size; (void)ws_size;
}
